// TransformerBlock_42056319762494
// MI455X (gfx1250) — matrix operand order repaired and checked by reading, not compiled, not run on hardware
//
#include <hip/hip_runtime.h>
#include <stdint.h>

typedef __attribute__((ext_vector_type(16))) __bf16 v16bf;
typedef __attribute__((ext_vector_type(8)))  __bf16 v8bf;
typedef __attribute__((ext_vector_type(8)))  float  v8f;

#define BATCH 32
#define NPOS  512
#define CH    512
#define NH    8
#define DH    64
#define FH    1024
#define BN_EPS 1e-5f

union ABFrag {
  v16bf v;
  v8bf  h[2];
  __bf16 e[16];
};

__device__ __forceinline__ v8f wmma_bf16(v16bf a, v16bf b, v8f c) {
  // D = A(16x32 bf16) x B(32x16 bf16) + C(16x16 f32)
  return __builtin_amdgcn_wmma_f32_16x16x32_bf16(false, a, false, b,
                                                 (short)0, c, false, false);
}

// CDNA5 async copy: 16 bytes global -> LDS per lane (ASYNCcnt tracked).
__device__ __forceinline__ void async_copy_b128(const __bf16* g, __bf16* l) {
  unsigned loff = (unsigned)(uintptr_t)l;                 // LDS byte offset
  unsigned long long ga = (unsigned long long)(uintptr_t)g;
  asm volatile("global_load_async_to_lds_b128 %0, %1, off"
               :: "v"(loff), "v"(ga) : "memory");
}
__device__ __forceinline__ void wait_asynccnt0() {
  asm volatile("s_wait_asynccnt 0x0" ::: "memory");
}

// ---------------------------------------------------------------------------
// Batched GEMM: Y[b] = W[M,K] @ X[b][K,512] + epilogue
//   EPI 0: + bias                      (YT = bf16)
//   EPI 1: + bias, exact GELU          (YT = bf16)
//   EPI 2: + bias, + residual, BN eval (YT = f32)
// XT = float (convert at stage) or __bf16 (raw stage).
// 256 threads / 8 waves, tile 128Mx128N, K-step 32, double-buffered LDS.
// ---------------------------------------------------------------------------
template <int EPI, typename XT, typename YT>
__global__ __launch_bounds__(256) void gemm_epi(
    const float* __restrict__ W, const XT* __restrict__ X,
    const float* __restrict__ bias, YT* __restrict__ Y,
    int M, int K,
    const float* __restrict__ res,
    const float* __restrict__ bng, const float* __restrict__ bnb,
    const float* __restrict__ bnm, const float* __restrict__ bnv)
{
  constexpr bool XB = (sizeof(XT) == 2);

  const int b    = blockIdx.z;
  const int m0   = blockIdx.y * 128;
  const int n0   = blockIdx.x * 128;
  const int tid  = threadIdx.x;
  const int lane = tid & 31;
  const int wv   = tid >> 5;
  const int l15  = lane & 15;
  const int half = lane >> 4;
  const int wm   = (wv & 3) * 32;
  const int wn   = (wv >> 2) * 64;

  const XT* Xb = X + (size_t)b * K * NPOS;
  YT*       Yb = Y + (size_t)b * M * NPOS;

  __shared__ __align__(16) __bf16 As[2][128][40];
  __shared__ __align__(16) __bf16 Bs[2][128][40];   // transposed: [n][k]

  float4 ra[4];
  float4 rbf[4];
  v8bf   rbh[2];

  auto loadA = [&](int k0) {
    #pragma unroll
    for (int r = 0; r < 4; ++r) {
      int id = tid + 256 * r, i = id >> 3, kq = (id & 7) * 4;
      ra[r] = *(const float4*)(W + (size_t)(m0 + i) * K + k0 + kq);
    }
  };
  auto storeA = [&](int bi) {
    #pragma unroll
    for (int r = 0; r < 4; ++r) {
      int id = tid + 256 * r, i = id >> 3, kq = (id & 7) * 4;
      As[bi][i][kq + 0] = (__bf16)ra[r].x; As[bi][i][kq + 1] = (__bf16)ra[r].y;
      As[bi][i][kq + 2] = (__bf16)ra[r].z; As[bi][i][kq + 3] = (__bf16)ra[r].w;
    }
  };
  auto loadB = [&](int k0) {
    if constexpr (XB) {
      #pragma unroll
      for (int r = 0; r < 2; ++r) {
        int id = tid + 256 * r, k = id >> 4, n8 = (id & 15) * 8;
        rbh[r] = *(const v8bf*)(Xb + (size_t)(k0 + k) * NPOS + n0 + n8);
      }
    } else {
      #pragma unroll
      for (int r = 0; r < 4; ++r) {
        int id = tid + 256 * r, k = id >> 5, nq = (id & 31) * 4;
        rbf[r] = *(const float4*)(Xb + (size_t)(k0 + k) * NPOS + n0 + nq);
      }
    }
  };
  auto storeB = [&](int bi) {
    if constexpr (XB) {
      #pragma unroll
      for (int r = 0; r < 2; ++r) {
        int id = tid + 256 * r, k = id >> 4, n8 = (id & 15) * 8;
        #pragma unroll
        for (int q = 0; q < 8; ++q) Bs[bi][n8 + q][k] = rbh[r][q];
      }
    } else {
      #pragma unroll
      for (int r = 0; r < 4; ++r) {
        int id = tid + 256 * r, k = id >> 5, nq = (id & 31) * 4;
        Bs[bi][nq + 0][k] = (__bf16)rbf[r].x; Bs[bi][nq + 1][k] = (__bf16)rbf[r].y;
        Bs[bi][nq + 2][k] = (__bf16)rbf[r].z; Bs[bi][nq + 3][k] = (__bf16)rbf[r].w;
      }
    }
  };

  v8f zero = {};
  v8f acc[2][4];
  #pragma unroll
  for (int i = 0; i < 2; ++i)
    #pragma unroll
    for (int j = 0; j < 4; ++j) acc[i][j] = zero;

  loadA(0); loadB(0);
  storeA(0); storeB(0);
  int cur = 0;

  for (int k0 = 0; k0 < K; k0 += 32) {
    __syncthreads();
    const bool more = (k0 + 32 < K);
    if (more) { loadA(k0 + 32); loadB(k0 + 32); }   // overlap with WMMAs

    ABFrag a[2], bf[4];
    #pragma unroll
    for (int ms = 0; ms < 2; ++ms) {
      const __bf16* ap = &As[cur][wm + ms * 16 + l15][half * 8];
      a[ms].h[0] = *(const v8bf*)ap;
      a[ms].h[1] = *(const v8bf*)(ap + 16);
    }
    #pragma unroll
    for (int js = 0; js < 4; ++js) {
      const __bf16* bp = &Bs[cur][wn + js * 16 + l15][half * 8];
      bf[js].h[0] = *(const v8bf*)bp;
      bf[js].h[1] = *(const v8bf*)(bp + 16);
    }
    #pragma unroll
    for (int ms = 0; ms < 2; ++ms)
      #pragma unroll
      for (int js = 0; js < 4; ++js)
        acc[ms][js] = wmma_bf16(a[ms].v, bf[js].v, acc[ms][js]);

    if (more) { storeA(cur ^ 1); storeB(cur ^ 1); }
    cur ^= 1;
  }

  // epilogue
  #pragma unroll
  for (int ms = 0; ms < 2; ++ms) {
    #pragma unroll
    for (int r = 0; r < 8; ++r) {
      const int mg = m0 + wm + ms * 16 + half * 8 + r;
      const float bi = bias[mg];
      float g = 0.f, bb = 0.f, mm = 0.f, iv = 0.f;
      if constexpr (EPI == 2) {
        g  = bng[mg]; bb = bnb[mg]; mm = bnm[mg];
        iv = __frsqrt_rn(bnv[mg] + BN_EPS);
      }
      #pragma unroll
      for (int js = 0; js < 4; ++js) {
        const int ng = n0 + wn + js * 16 + l15;
        float v = acc[ms][js][r] + bi;
        if constexpr (EPI == 1) {
          v = 0.5f * v * (1.0f + erff(v * 0.70710678118654752f));
        } else if constexpr (EPI == 2) {
          v += res[(size_t)b * M * NPOS + (size_t)mg * NPOS + ng];
          v = (v - mm) * (g * iv) + bb;
        }
        Yb[(size_t)mg * NPOS + ng] = (YT)v;
      }
    }
  }
}

// ---------------------------------------------------------------------------
// Attention: per (b, h, 64-query block); 128 threads = 4 waves. qkv is bf16.
// Big-LDS (CDNA5 320 KB/WGP):
//   S     : f32 scores [64][516] (132096 B) (rows rewritten in-place as bf16
//                                            probs by softmax; head reused as
//                                            bf16 out-stage at the end)
//   Klds  : bf16 K     [512][72] ( 73728 B) layout [m][d]
//   VT    : bf16 V     [64][520] ( 66560 B) layout [d][m], async-copied
//   Qlds  : bf16 Q     [64][72]  (  9216 B) layout [n][d]
//   BiasL : f32 rpe slice [576]  (  2304 B) 575 consecutive entries, no clip
// ---------------------------------------------------------------------------
#define S_STRIDE 516
#define K_STRIDE 72
#define V_STRIDE 520
#define Q_STRIDE 72
#define O_STRIDE 80
#define ATTN_LDS_BYTES (64*S_STRIDE*4 + 512*K_STRIDE*2 + 64*V_STRIDE*2 + 64*Q_STRIDE*2 + 576*4)

__global__ __launch_bounds__(128) void attn_kernel(
    const __bf16* __restrict__ qkv,  // [B, 3*CH, NPOS] bf16
    const float*  __restrict__ rpe,  // [1023, NH]
    __bf16* __restrict__ out)        // [B, CH, NPOS] bf16
{
  const int b    = blockIdx.z;
  const int h    = blockIdx.y;
  const int q0   = blockIdx.x * 64;
  const int tid  = threadIdx.x;
  const int lane = tid & 31;
  const int wv   = tid >> 5;
  const int l15  = lane & 15;
  const int half = lane >> 4;

  extern __shared__ __align__(16) char smem[];
  float*  S     = (float*)smem;
  __bf16* Klds  = (__bf16*)(smem + (size_t)64 * S_STRIDE * 4);
  __bf16* VT    = Klds + (size_t)512 * K_STRIDE;
  __bf16* Qlds  = VT + (size_t)64 * V_STRIDE;
  float*  BiasL = (float*)(Qlds + (size_t)64 * Q_STRIDE);

  const size_t base = ((size_t)b * (3 * CH) + (size_t)h * DH) * NPOS;
  const __bf16* Qg = qkv + base;
  const __bf16* Kg = qkv + base + (size_t)CH * NPOS;
  const __bf16* Vg = qkv + base + (size_t)2 * CH * NPOS;

  // ---- V: pure byte copy -> CDNA5 async global->LDS (ASYNCcnt)
  #pragma unroll 4
  for (int r = 0; r < 32; ++r) {
    int id = tid + 128 * r;          // 4096 x 16B chunks
    int d  = id >> 6;
    int m8 = (id & 63) * 8;
    async_copy_b128(Vg + (size_t)d * NPOS + m8, VT + (size_t)d * V_STRIDE + m8);
  }
  // ---- rel-pos bias slice: indices 448-q0 .. 1022-q0 (always in range)
  for (int t = tid; t < 575; t += 128)
    BiasL[t] = rpe[(448 - q0 + t) * NH + h];
  // ---- Q tile (64d x 64n) -> Qlds[n][d]  (transpose scatter, no cvt)
  #pragma unroll
  for (int r = 0; r < 4; ++r) {
    int id = tid + 128 * r;          // 512 x 8-elt chunks
    int d  = id >> 3;
    int n8 = (id & 7) * 8;
    v8bf v = *(const v8bf*)(Qg + (size_t)d * NPOS + q0 + n8);
    #pragma unroll
    for (int q = 0; q < 8; ++q) Qlds[(n8 + q) * Q_STRIDE + d] = v[q];
  }
  // ---- K (64d x 512m) -> Klds[m][d]  (transpose scatter, no cvt)
  #pragma unroll 4
  for (int r = 0; r < 32; ++r) {
    int id = tid + 128 * r;          // 4096 x 8-elt chunks
    int d  = id >> 6;
    int m8 = (id & 63) * 8;
    v8bf v = *(const v8bf*)(Kg + (size_t)d * NPOS + m8);
    #pragma unroll
    for (int q = 0; q < 8; ++q) Klds[(m8 + q) * K_STRIDE + d] = v[q];
  }
  wait_asynccnt0();
  __syncthreads();

  // ---- phase 1: S = Q K^T   (scale/bias folded into softmax read)
  const int qr0 = wv * 16;
  ABFrag aQ[2];
  #pragma unroll
  for (int kk = 0; kk < 2; ++kk) {
    const __bf16* qp = &Qlds[(qr0 + l15) * Q_STRIDE + kk * 32 + half * 8];
    aQ[kk].h[0] = *(const v8bf*)qp;
    aQ[kk].h[1] = *(const v8bf*)(qp + 16);
  }
  v8f zero = {};
  for (int j = 0; j < 32; ++j) {
    ABFrag bK0, bK1;
    const __bf16* kp = &Klds[(j * 16 + l15) * K_STRIDE + half * 8];
    bK0.h[0] = *(const v8bf*)kp;         bK0.h[1] = *(const v8bf*)(kp + 16);
    bK1.h[0] = *(const v8bf*)(kp + 32);  bK1.h[1] = *(const v8bf*)(kp + 48);
    v8f sc = wmma_bf16(aQ[0].v, bK0.v, zero);
    sc = wmma_bf16(aQ[1].v, bK1.v, sc);
    float* Sc = &S[(qr0 + half * 8) * S_STRIDE + j * 16 + l15];
    #pragma unroll
    for (int r = 0; r < 8; ++r) Sc[r * S_STRIDE] = sc[r];
  }

  // ---- phase 2: softmax with fused scale+bias; write bf16 probs in place
  const float scale = 0.125f;  // Dh^-0.5
  for (int rr = 0; rr < 16; ++rr) {
    const int srow = qr0 + rr;
    float* Srow = &S[(size_t)srow * S_STRIDE];
    const float* Bl = BiasL + 63 - srow;
    float vals[16];
    float mx = -1e30f;
    #pragma unroll
    for (int i = 0; i < 16; ++i) {
      const int c = lane + 32 * i;
      vals[i] = fmaf(Srow[c], scale, Bl[c]);
      mx = fmaxf(mx, vals[i]);
    }
    #pragma unroll
    for (int off = 16; off; off >>= 1) mx = fmaxf(mx, __shfl_xor(mx, off, 32));
    float sum = 0.f;
    #pragma unroll
    for (int i = 0; i < 16; ++i) { vals[i] = __expf(vals[i] - mx); sum += vals[i]; }
    #pragma unroll
    for (int off = 16; off; off >>= 1) sum += __shfl_xor(sum, off, 32);
    const float inv = __frcp_rn(sum);
    __bf16* Prow = (__bf16*)Srow;          // safe: row fully read above
    #pragma unroll
    for (int i = 0; i < 16; ++i) Prow[lane + 32 * i] = (__bf16)(vals[i] * inv);
  }

  // ---- phase 3: out = P @ V   (P read directly as bf16 fragments)
  v8f acc4[4];
  #pragma unroll
  for (int dt = 0; dt < 4; ++dt) acc4[dt] = zero;

  const __bf16* Pme = (const __bf16*)&S[(size_t)(qr0 + l15) * S_STRIDE];
  for (int kk = 0; kk < 512; kk += 32) {
    ABFrag aP;
    aP.h[0] = *(const v8bf*)(Pme + kk + half * 8);
    aP.h[1] = *(const v8bf*)(Pme + kk + 16 + half * 8);
    #pragma unroll
    for (int dt = 0; dt < 4; ++dt) {
      ABFrag bV;
      const __bf16* vp = &VT[(dt * 16 + l15) * V_STRIDE + kk + half * 8];
      bV.h[0] = *(const v8bf*)vp;
      bV.h[1] = *(const v8bf*)(vp + 16);
      acc4[dt] = wmma_bf16(aP.v, bV.v, acc4[dt]);
    }
  }

  // ---- phase 4: stage bf16 output through LDS, exit coalesced b128
  __syncthreads();                          // all waves done reading S/P
  __bf16* Obuf = (__bf16*)smem;             // [64 d][80] bf16, reuses S head
  #pragma unroll
  for (int dt = 0; dt < 4; ++dt) {
    const int d = dt * 16 + l15;
    #pragma unroll
    for (int r = 0; r < 8; ++r)
      Obuf[d * O_STRIDE + qr0 + half * 8 + r] = (__bf16)acc4[dt][r];
  }
  __syncthreads();
  __bf16* outb = out + ((size_t)b * CH + (size_t)h * DH) * NPOS;
  #pragma unroll
  for (int r = 0; r < 4; ++r) {
    int id = tid + 128 * r;                 // 512 x 8-elt chunks
    int d  = id >> 3;
    int c8 = (id & 7) * 8;
    v8bf vv = *(const v8bf*)(Obuf + d * O_STRIDE + c8);
    *(v8bf*)(outb + (size_t)d * NPOS + q0 + c8) = vv;
  }
}

// ---------------------------------------------------------------------------
extern "C" void kernel_launch(void* const* d_in, const int* in_sizes, int n_in,
                              void* d_out, int out_size, void* d_ws, size_t ws_size,
                              hipStream_t stream) {
  (void)in_sizes; (void)n_in; (void)out_size; (void)ws_size;
  const float* x      = (const float*)d_in[0];
  const float* qkv_w  = (const float*)d_in[1];
  const float* qkv_b  = (const float*)d_in[2];
  const float* out_w  = (const float*)d_in[3];
  const float* out_b  = (const float*)d_in[4];
  const float* rpe    = (const float*)d_in[5];
  const float* bn1_g  = (const float*)d_in[6];
  const float* bn1_b  = (const float*)d_in[7];
  const float* bn1_m  = (const float*)d_in[8];
  const float* bn1_v  = (const float*)d_in[9];
  const float* ffn1_w = (const float*)d_in[10];
  const float* ffn1_b = (const float*)d_in[11];
  const float* ffn2_w = (const float*)d_in[12];
  const float* ffn2_b = (const float*)d_in[13];
  const float* bn2_g  = (const float*)d_in[14];
  const float* bn2_b  = (const float*)d_in[15];
  const float* bn2_m  = (const float*)d_in[16];
  const float* bn2_v  = (const float*)d_in[17];

  char* ws = (char*)d_ws;
  __bf16* qkv_buf  = (__bf16*)ws;  ws += (size_t)BATCH * 3 * CH * NPOS * 2;
  __bf16* attn_out = (__bf16*)ws;  ws += (size_t)BATCH * CH * NPOS * 2;
  float*  h_buf    = (float*)ws;   ws += (size_t)BATCH * CH * NPOS * 4;
  __bf16* f_buf    = (__bf16*)ws;

  const dim3 blk(256);

  // 1) qkv projection -> bf16
  gemm_epi<0, float, __bf16><<<dim3(4, 12, BATCH), blk, 0, stream>>>(
      qkv_w, x, qkv_b, qkv_buf, 3 * CH, CH,
      nullptr, nullptr, nullptr, nullptr, nullptr);

  // 2) attention -> bf16
  attn_kernel<<<dim3(NPOS / 64, NH, BATCH), dim3(128), ATTN_LDS_BYTES, stream>>>(
      qkv_buf, rpe, attn_out);

  // 3) out projection + residual(x) + BN1 -> h (f32)
  gemm_epi<2, __bf16, float><<<dim3(4, 4, BATCH), blk, 0, stream>>>(
      out_w, attn_out, out_b, h_buf, CH, CH,
      x, bn1_g, bn1_b, bn1_m, bn1_v);

  // 4) FFN1 + GELU -> f (bf16)
  gemm_epi<1, float, __bf16><<<dim3(4, 8, BATCH), blk, 0, stream>>>(
      ffn1_w, h_buf, ffn1_b, f_buf, FH, CH,
      nullptr, nullptr, nullptr, nullptr, nullptr);

  // 5) FFN2 + residual(h) + BN2 -> out (f32)
  gemm_epi<2, __bf16, float><<<dim3(4, 4, BATCH), blk, 0, stream>>>(
      ffn2_w, f_buf, ffn2_b, (float*)d_out, CH, FH,
      h_buf, bn2_g, bn2_b, bn2_m, bn2_v);
}
